// KBGraphAttentionalHead_71459665871400
// MI455X (gfx1250) — matrix operand order repaired and checked by reading, not compiled, not run on hardware
//
#include <hip/hip_runtime.h>

typedef __attribute__((ext_vector_type(16))) _Float16 v16h;
typedef __attribute__((ext_vector_type(8)))  float    v8f;

#define N_NODES 50000
#define N_EDGES 800000
#define IN_DIM  192
#define D_OUT   64

// ---- order-preserving float<->uint encoding for atomicMax on floats ----
__device__ __forceinline__ unsigned encf(float x) {
  unsigned u = __float_as_uint(x);
  return (u & 0x80000000u) ? ~u : (u | 0x80000000u);
}
__device__ __forceinline__ float decf(unsigned u) {
  return (u & 0x80000000u) ? __uint_as_float(u & 0x7FFFFFFFu)
                           : __uint_as_float(~u);
}

__device__ __forceinline__ float mishf(float x) {
  // numerically stable softplus
  float sp = (x > 0.0f) ? (x + log1pf(expf(-x))) : log1pf(expf(x));
  return x * tanhf(sp);
}

// ---------------- K0: zero output, denom, node-max ----------------
__global__ void k0_init(float* __restrict__ out, float* __restrict__ denom,
                        unsigned* __restrict__ nmax) {
  int i = blockIdx.x * blockDim.x + threadIdx.x;
  if (i < N_NODES * D_OUT) out[i] = 0.0f;
  if (i < N_NODES) { denom[i] = 0.0f; nmax[i] = 0u; }
}

// ---------------- K1: WMMA GEMM + attention logit -----------------
// One wave per 16-edge tile. 4 N-tiles of 16 => full D_OUT=64.
// K = 192 = 6 steps of 32 (f16 WMMA 16x16x32, fp32 accumulate).
__global__ __launch_bounds__(256) void k1_gemm(
    const float* __restrict__ tf,          // [E, 192]
    const long long* __restrict__ dstp,    // indices row 0: dst per edge
    const float* __restrict__ W,           // [64, 192] row-major
    const float* __restrict__ watt,        // [64]
    float* __restrict__ h_ws,              // [E, 64]
    float* __restrict__ a_ws,              // [E]
    unsigned* __restrict__ nmax)           // [N] encoded max
{
  const int lane = threadIdx.x & 31;
  const int wave = threadIdx.x >> 5;
  const int mt   = blockIdx.x * 8 + wave;   // 0..49999 exactly
  const int eb   = mt * 16;                 // first edge of tile
  const int n    = lane & 15;               // row (A) / col (B,C,D) id
  const int hi   = lane >> 4;               // half-wave select

  v8f acc[4] = {v8f{}, v8f{}, v8f{}, v8f{}};

#pragma unroll
  for (int kt = 0; kt < 6; ++kt) {
    // ---- A fragment: 16x32 f16. Lane holds row m=n; K layout per ISA:
    // v16h elems j=0..7  -> K = kt*32 + 8*hi + j
    //           j=8..15 -> K = kt*32 + 8*hi + 8 + j
    const float* rA = tf + (size_t)(eb + n) * IN_DIM + kt * 32 + 8 * hi;
    const float4* pA = (const float4*)rA;
    float4 A0 = pA[0], A1 = pA[1], A2 = pA[4], A3 = pA[5];
    v16h af;
    af[0]  = (_Float16)A0.x; af[1]  = (_Float16)A0.y;
    af[2]  = (_Float16)A0.z; af[3]  = (_Float16)A0.w;
    af[4]  = (_Float16)A1.x; af[5]  = (_Float16)A1.y;
    af[6]  = (_Float16)A1.z; af[7]  = (_Float16)A1.w;
    af[8]  = (_Float16)A2.x; af[9]  = (_Float16)A2.y;
    af[10] = (_Float16)A2.z; af[11] = (_Float16)A2.w;
    af[12] = (_Float16)A3.x; af[13] = (_Float16)A3.y;
    af[14] = (_Float16)A3.z; af[15] = (_Float16)A3.w;

#pragma unroll
    for (int t = 0; t < 4; ++t) {
      // ---- B fragment: 32x16 f16, B[k][nc] = W[nc][k].
      // Lane holds col nc = t*16+n ; v16h elem j -> K = kt*32 + 8*hi + j (j<8), kt*32 + 8*hi + 8 + j (j>=8)
      const float* rB = W + (size_t)(t * 16 + n) * IN_DIM + kt * 32 + 8 * hi;
      const float4* pB = (const float4*)rB;
      float4 B0 = pB[0], B1 = pB[1], B2 = pB[4], B3 = pB[5];
      v16h bf;
      bf[0]  = (_Float16)B0.x; bf[1]  = (_Float16)B0.y;
      bf[2]  = (_Float16)B0.z; bf[3]  = (_Float16)B0.w;
      bf[4]  = (_Float16)B1.x; bf[5]  = (_Float16)B1.y;
      bf[6]  = (_Float16)B1.z; bf[7]  = (_Float16)B1.w;
      bf[8]  = (_Float16)B2.x; bf[9]  = (_Float16)B2.y;
      bf[10] = (_Float16)B2.z; bf[11] = (_Float16)B2.w;
      bf[12] = (_Float16)B3.x; bf[13] = (_Float16)B3.y;
      bf[14] = (_Float16)B3.z; bf[15] = (_Float16)B3.w;

      acc[t] = __builtin_amdgcn_wmma_f32_16x16x32_f16(
          /*neg_a=*/false, af, /*neg_b=*/false, bf,
          /*c_mod=*/(short)0, acc[t], /*reuse_a=*/false, /*reuse_b=*/false);
    }
  }

  // ---- attention logit: dot(h_row, w_att) via in-register + shfl reduce.
  // C/D layout: lane holds col n, rows m = r + 8*hi for VGPR r.
  float wv0 = watt[n], wv1 = watt[16 + n], wv2 = watt[32 + n], wv3 = watt[48 + n];
  float dotv[8];
#pragma unroll
  for (int r = 0; r < 8; ++r) {
    float s = acc[0][r] * wv0 + acc[1][r] * wv1 + acc[2][r] * wv2 + acc[3][r] * wv3;
    s += __shfl_xor(s, 1, 32);
    s += __shfl_xor(s, 2, 32);
    s += __shfl_xor(s, 4, 32);
    s += __shfl_xor(s, 8, 32);   // xor<16: stays within the 16-lane half
    dotv[r] = s;
  }

  // ---- store h tile
#pragma unroll
  for (int r = 0; r < 8; ++r) {
    int m = r + 8 * hi;
    size_t base = (size_t)(eb + m) * D_OUT + n;
    h_ws[base]      = acc[0][r];
    h_ws[base + 16] = acc[1][r];
    h_ws[base + 32] = acc[2][r];
    h_ws[base + 48] = acc[3][r];
  }

  // ---- logits + segment max (one lane per half-wave)
  if (n == 0) {
#pragma unroll
    for (int r = 0; r < 8; ++r) {
      int m = r + 8 * hi;
      float av = mishf(dotv[r]);
      a_ws[eb + m] = av;
      int d = (int)dstp[eb + m];
      atomicMax(&nmax[d], encf(av));
    }
  }
}

// ---------------- K2: e = exp(a - max[dst]); denom[dst] += e ----------------
__global__ void k2_exp(const long long* __restrict__ dstp,
                       const float* __restrict__ a_ws,
                       const unsigned* __restrict__ nmax,
                       float* __restrict__ e_ws,
                       float* __restrict__ denom) {
  int e = blockIdx.x * blockDim.x + threadIdx.x;
  if (e >= N_EDGES) return;
  int d = (int)dstp[e];
  float mx = decf(nmax[d]);
  float ex = expf(a_ws[e] - mx);
  e_ws[e] = ex;
  atomicAdd(&denom[d], ex);
}

// ---------------- K3: out[dst] += h * e/denom[dst] ----------------
__global__ void k3_scatter(const long long* __restrict__ dstp,
                           const float* __restrict__ h_ws,
                           const float* __restrict__ e_ws,
                           const float* __restrict__ denom,
                           float* __restrict__ out) {
  unsigned i = blockIdx.x * blockDim.x + threadIdx.x;  // < E*64
  unsigned e = i >> 6;
  unsigned dim = i & 63u;
  int d = (int)dstp[e];
  float p = e_ws[e] / denom[d];
  atomicAdd(&out[(size_t)d * D_OUT + dim], h_ws[i] * p);
}

extern "C" void kernel_launch(void* const* d_in, const int* in_sizes, int n_in,
                              void* d_out, int out_size, void* d_ws, size_t ws_size,
                              hipStream_t stream) {
  const float*     tf   = (const float*)d_in[0];
  const long long* idx  = (const long long*)d_in[1];  // [2, E] int64; row 0 = dst
  const float*     W    = (const float*)d_in[2];
  const float*     watt = (const float*)d_in[3];
  float* out = (float*)d_out;

  char* ws = (char*)d_ws;
  float*    h_ws  = (float*)ws;                               // E*64 floats
  float*    a_ws  = (float*)(ws + (size_t)N_EDGES * D_OUT * 4);
  float*    e_ws  = a_ws + N_EDGES;
  float*    denom = e_ws + N_EDGES;
  unsigned* nmax  = (unsigned*)(denom + N_NODES);

  const long long* dstp = idx;  // first E elements are the dst row

  // K0: init out/denom/nmax
  {
    int n = N_NODES * D_OUT;
    k0_init<<<(n + 255) / 256, 256, 0, stream>>>(out, denom, nmax);
  }
  // K1: WMMA GEMM + logits + segment max (50000 waves, 8 waves/block)
  k1_gemm<<<6250, 256, 0, stream>>>(tf, dstp, W, watt, h_ws, a_ws, nmax);
  // K2: exp + denom
  k2_exp<<<(N_EDGES + 255) / 256, 256, 0, stream>>>(dstp, a_ws, nmax, e_ws, denom);
  // K3: weighted scatter-sum (E*64 threads)
  k3_scatter<<<(N_EDGES * 64) / 256, 256, 0, stream>>>(dstp, h_ws, e_ws, denom, out);
}
